// CTRSelfAttention_60696477827086
// MI455X (gfx1250) — compile-verified, matrix operand order checked by reading, not run on hardware
//
#include <hip/hip_runtime.h>
#include <hip/hip_bf16.h>

typedef __attribute__((ext_vector_type(16))) __bf16        v16bf;
typedef __attribute__((ext_vector_type(8)))  float         v8f;
typedef __attribute__((ext_vector_type(8)))  unsigned int  v8u;

constexpr int Bc = 32;    // batch
constexpr int Lc = 1024;  // sequence length
constexpr int Ec = 512;   // embed dim
constexpr int Oc = 512;   // output dim (head dim)

// float -> bf16 bits via native fptrunc (backend selects hw bf16 convert)
__device__ __forceinline__ unsigned short bfbits(float f) {
  __bf16 b = (__bf16)f;
  return __builtin_bit_cast(unsigned short, b);
}

__device__ __forceinline__ unsigned int packbf2(float x, float y) {
  return (unsigned int)bfbits(x) | ((unsigned int)bfbits(y) << 16);
}

// ISA 16-bit A/B matrix VGPR layout: VGPR v (0..7), lane-half h (0/1) holds
// the K pair starting at this index (K and K+1 packed in one dword).
__device__ __forceinline__ int kpair_base(int v, int h) {
  return (v < 4 ? (v << 1) : 16 + ((v - 4) << 1)) + (h << 3);
}

__device__ __forceinline__ v8f wmma_bf16(const v8u& a, const v8u& b, v8f c) {
  return __builtin_amdgcn_wmma_f32_16x16x32_bf16(
      /*neg_a=*/false, __builtin_bit_cast(v16bf, a),
      /*neg_b=*/false, __builtin_bit_cast(v16bf, b),
      /*c_mod=*/(short)0, c, /*reuse_a=*/false, /*reuse_b=*/false);
}

// ---------------------------------------------------------------------------
// Kernel 1: QKV projection  [B*L, E] x [E, O] -> bf16 Q/K in row-major,
// V transposed [B][O][L] so the PV B-operand has contiguous K pairs.
// grid = (O/64, B*L/64, 3), block = 128. Each wave owns a 16-col stripe and
// reuses one B tile across 4 row tiles (4 back-to-back WMMAs per k-step).
// ---------------------------------------------------------------------------
__global__ __launch_bounds__(128) void qkv_proj_kernel(
    const float* __restrict__ J,
    const float* __restrict__ Wq, const float* __restrict__ Wk,
    const float* __restrict__ Wv,
    unsigned short* __restrict__ Qws, unsigned short* __restrict__ Kws,
    unsigned short* __restrict__ Vt) {
  const int zsel = blockIdx.z;
  const float* W = (zsel == 0) ? Wq : (zsel == 1) ? Wk : Wv;
  unsigned short* Out = (zsel == 0) ? Qws : (zsel == 1) ? Kws : Vt;

  const int lane = threadIdx.x & 31;
  const int wave = threadIdx.x >> 5;
  const int h    = lane >> 4;
  const int mn   = lane & 15;
  const int row0 = blockIdx.y << 6;
  const int col0 = (blockIdx.x << 6) + (wave << 4);

  v8f acc[4] = {{}, {}, {}, {}};
  for (int e0 = 0; e0 < Ec; e0 += 32) {
    v8u bu;
#pragma unroll
    for (int v = 0; v < 8; ++v) {
      const int kb = kpair_base(v, h);
      bu[v] = packbf2(W[(size_t)(e0 + kb) * Oc + col0 + mn],
                      W[(size_t)(e0 + kb + 1) * Oc + col0 + mn]);
    }
#pragma unroll
    for (int rt = 0; rt < 4; ++rt) {
      const float* arow = J + (size_t)(row0 + (rt << 4) + mn) * Ec + e0;
      v8u au;
#pragma unroll
      for (int v = 0; v < 8; ++v) {
        const int kb = kpair_base(v, h);
        const float2 f = *(const float2*)(arow + kb);
        au[v] = packbf2(f.x, f.y);
      }
      acc[rt] = wmma_bf16(au, bu, acc[rt]);
    }
  }
#pragma unroll
  for (int rt = 0; rt < 4; ++rt) {
#pragma unroll
    for (int v = 0; v < 8; ++v) {
      const int m    = v + (h << 3);
      const int grow = row0 + (rt << 4) + m;           // global row = b*L + l
      if (zsel == 2) {                                 // V: transposed [B][O][L]
        const int bb = grow >> 10;                     // L == 1024
        const int ll = grow & (Lc - 1);
        Out[((size_t)bb * Oc + col0 + mn) * Lc + ll] = bfbits(acc[rt][v]);
      } else {                                         // Q/K: row-major [B*L][O]
        Out[(size_t)grow * Oc + col0 + mn] = bfbits(acc[rt][v]);
      }
    }
  }
}

// ---------------------------------------------------------------------------
// Kernel 2: fused scores(QK^T)+delta -> softmax -> post-mask -> PV.
// grid = (L/16, B), block = 128 (4 waves). One (batch, 16-query) stripe per
// workgroup; full 1024-key score row lives in LDS (never touches HBM).
// ---------------------------------------------------------------------------
__global__ __launch_bounds__(128) void attn_kernel(
    const unsigned short* __restrict__ Qws,
    const unsigned short* __restrict__ Kws,
    const unsigned short* __restrict__ Vt,   // [B][O][L] bf16
    const float* __restrict__ delta,         // [B, L, L, 4]
    const int* __restrict__ traj,            // [B]
    float* __restrict__ out) {               // [B, L, O]
  __shared__ float        sS[16 * Lc];        // 64 KB raw scores / exp values
  __shared__ unsigned int sP[16 * (Lc / 2)];  // 32 KB packed bf16 probs
  __shared__ float        sRed[16][8];

  const int b    = blockIdx.y;
  const int q0   = blockIdx.x << 4;
  const int tid  = threadIdx.x;
  const int lane = tid & 31;
  const int wave = tid >> 5;
  const int h    = lane >> 4;
  const int mn   = lane & 15;

  // ---- Phase 1: scores = Q K^T + sum_k(delta), waves split the key tiles ---
  const unsigned short* qrow = Qws + (size_t)(b * Lc + q0 + mn) * Oc;
  for (int kt = wave; kt < Lc / 16; kt += 4) {
    const int k0 = kt << 4;
    // prefetch this wave's next delta tile (global_prefetch_b8)
    if (k0 + 64 < Lc) {
#pragma unroll
      for (int v = 0; v < 8; ++v) {
        const int m = v + (h << 3);
        __builtin_prefetch(
            delta + ((((size_t)b * Lc + (q0 + m)) * Lc) + (k0 + 64 + mn)) * 4,
            0, 1);
      }
    }
    const unsigned short* krow = Kws + (size_t)(b * Lc + k0 + mn) * Oc;
    v8f acc = {};
    for (int e0 = 0; e0 < Ec; e0 += 32) {
      v8u au, bu;
#pragma unroll
      for (int v = 0; v < 8; ++v) {
        const int kb = kpair_base(v, h);
        au[v] = *(const unsigned int*)(qrow + e0 + kb);  // Q[m][e..e+1]
        bu[v] = *(const unsigned int*)(krow + e0 + kb);  // K[n][e..e+1]
      }
      acc = wmma_bf16(au, bu, acc);
    }
#pragma unroll
    for (int v = 0; v < 8; ++v) {
      const int m = v + (h << 3);
      const float4 d4 = *(const float4*)(
          delta + ((((size_t)b * Lc + (q0 + m)) * Lc) + (k0 + mn)) * 4);
      sS[m * Lc + k0 + mn] = acc[v] + d4.x + d4.y + d4.z + d4.w;
    }
  }
  __syncthreads();

  // ---- Phase 2: softmax over full key row, then post-softmax mask ----------
  const int r   = tid >> 3;  // query row 0..15
  const int sub = tid & 7;   // 8 threads per row
  float pm = -3.402823e38f;
  for (int c = sub; c < Lc; c += 8) pm = fmaxf(pm, sS[r * Lc + c]);
  sRed[r][sub] = pm;
  __syncthreads();
  float rmax = sRed[r][0];
#pragma unroll
  for (int j = 1; j < 8; ++j) rmax = fmaxf(rmax, sRed[r][j]);
  __syncthreads();
  float ps = 0.f;
  for (int c = sub; c < Lc; c += 8) {
    const float e = __expf(sS[r * Lc + c] - rmax);
    sS[r * Lc + c] = e;
    ps += e;
  }
  sRed[r][sub] = ps;
  __syncthreads();
  float rsum = 0.f;
#pragma unroll
  for (int j = 0; j < 8; ++j) rsum += sRed[r][j];
  const float inv = 1.0f / rsum;  // denominator over ALL keys (mask is post-softmax)
  const int  t  = traj[b];
  const bool qv = (q0 + r) < t;
  for (int j2 = sub; j2 < Lc / 2; j2 += 8) {
    const int c0 = j2 << 1;
    const float p0 = (qv && (c0 < t))     ? sS[r * Lc + c0] * inv     : 0.f;
    const float p1 = (qv && (c0 + 1 < t)) ? sS[r * Lc + c0 + 1] * inv : 0.f;
    sP[r * (Lc / 2) + j2] = packbf2(p0, p1);
  }
  __syncthreads();

  // ---- Phase 3: out = P @ V^T-layout, waves split output-column tiles ------
  const unsigned int* prow = sP + mn * (Lc / 2);
  for (int nt = wave; nt < Oc / 16; nt += 4) {
    const int col0 = nt << 4;
    // V^T row for this lane's output column: contiguous in k
    const unsigned short* vtrow = Vt + (size_t)(b * Oc + col0 + mn) * Lc;
    v8f acc = {};
    for (int k0 = 0; k0 < Lc; k0 += 32) {
      v8u au, bu;
#pragma unroll
      for (int v = 0; v < 8; ++v) {
        const int kb = kpair_base(v, h);
        au[v] = prow[(k0 + kb) >> 1];                          // ds_load_b32
        bu[v] = *(const unsigned int*)(vtrow + k0 + kb);       // V[k..k+1][n]
      }
      acc = wmma_bf16(au, bu, acc);
    }
#pragma unroll
    for (int v = 0; v < 8; ++v) {
      const int m = v + (h << 3);
      out[((size_t)b * Lc + q0 + m) * Oc + col0 + mn] = acc[v];
    }
  }
}

// ---------------------------------------------------------------------------
extern "C" void kernel_launch(void* const* d_in, const int* in_sizes, int n_in,
                              void* d_out, int out_size, void* d_ws, size_t ws_size,
                              hipStream_t stream) {
  const float* J     = (const float*)d_in[0];
  const float* delta = (const float*)d_in[1];
  const float* Wq    = (const float*)d_in[2];
  const float* Wk    = (const float*)d_in[3];
  const float* Wv    = (const float*)d_in[4];
  const int*   traj  = (const int*)d_in[5];
  float*       out   = (float*)d_out;

  const size_t nQKV = (size_t)Bc * Lc * Oc;
  unsigned short* Qws = (unsigned short*)d_ws;
  unsigned short* Kws = Qws + nQKV;
  unsigned short* Vt  = Kws + nQKV;

  dim3 g1(Oc / 64, (Bc * Lc) / 64, 3);
  qkv_proj_kernel<<<g1, 128, 0, stream>>>(J, Wq, Wk, Wv, Qws, Kws, Vt);

  dim3 g2(Lc / 16, Bc);
  attn_kernel<<<g2, 128, 0, stream>>>(Qws, Kws, Vt, delta, traj, out);
}
